// CEBlock_2774548873279
// MI455X (gfx1250) — matrix operand order repaired and checked by reading, not compiled, not run on hardware
//
#include <hip/hip_runtime.h>
#include <hip/hip_bf16.h>

// ---------------------------------------------------------------------------
// Fused Clifford-algebra block for MI455X (gfx1250, wave32, WMMA).
//   qkv GEMM  : bf16 v_wmma_f32_16x16x32_bf16, f32 accumulate, K=128
//   weights   : pass A -> global_load_async_to_lds_b128 (ASYNCcnt)
//               pass B -> Tensor Data Mover tensor_load_to_lds (TENSORcnt)
//               both double-buffered raw fp32 -> bf16 fragment layout in LDS
//   epilogue  : SiLU gating, v-grade-norm, geometric product, cross-channel
//               norm, fused.  Two passes over channels (norm coupling);
//               A-fragments live in registers so the k/v recompute only costs
//               tensor-core cycles, not memory traffic.
// ---------------------------------------------------------------------------

typedef __bf16 bf16_t;
typedef bf16_t v16bf __attribute__((ext_vector_type(16)));
typedef float  v8f   __attribute__((ext_vector_type(8)));

#define B_TOTAL  32768
#define CIN      128
#define COUT     128
#define BM       64            // batch rows per workgroup
#define THREADS  128           // 4 waves (wave32)
#define MPITCH   136           // bf16 pitch along K (128 + 8 pad, keeps 16B align)
#define LX_HALFS (8 * BM * MPITCH)   // 69632 halves = 136 KB shared pool
#define RAW0_H   26112               // lwbf occupies [0, 26112) halves (12*16*136)
#define RAW1_H   (26112 + 16384)     // each raw buffer = 32 KB = 16384 halves
#define EPS_     1e-6f
#define RSQRT2_  0.70710678118654752f

// ---- Clifford Cl(3,0) compile-time tables --------------------------------
__host__ __device__ constexpr int bmask(int i){ constexpr int M[8]={0,1,2,4,3,5,6,7}; return M[i]; }
__host__ __device__ constexpr int bidx (int m){ constexpr int M[8]={0,1,2,4,3,5,6,7}; return M[m]; }
__host__ __device__ constexpr int bgrade(int i){ constexpr int G[8]={0,1,1,1,2,2,2,3}; return G[i]; }
__host__ __device__ constexpr int gpOut(int i,int k){ return bidx(bmask(i)^bmask(k)); }
__host__ __device__ constexpr float gpSign(int i,int k){
  int a=bmask(i), b=bmask(k), c=0;
  for(int p=0;p<3;++p) for(int q=0;q<3;++q)
    if(((a>>p)&1) && ((b>>q)&1) && (p>q)) ++c;
  return (c&1) ? -1.0f : 1.0f;
}

// ---- helpers -------------------------------------------------------------
__device__ __forceinline__ unsigned int pack_bf16x2(float lo, float hi){
  unsigned int a = __float_as_uint(lo), b = __float_as_uint(hi);
  a = (a + 0x7FFFu + ((a>>16)&1u)) >> 16;            // RNE
  b = (b + 0x7FFFu + ((b>>16)&1u)) >> 16;
  return (a & 0xFFFFu) | (b << 16);
}
__device__ __forceinline__ float sigm(float x){ return 1.0f/(1.0f + __expf(-x)); }
__device__ __forceinline__ unsigned lds_off(const void* p){
  // generic pointer to LDS: low 32 bits are the LDS-space byte address
  return (unsigned)(unsigned long long)p;
}
#define WAIT_ASYNC(n) asm volatile("s_wait_asynccnt " #n ::: "memory")

union Frag { uint4 u[2]; v16bf v; };

// ---- async raw-weight staging (GVS: SGPR base + VGPR byte offset) --------
// copies one 32 KB chunk (16 channels x 128 K x 4 grades, fp32) into LDS
__device__ __forceinline__ void async_copy_chunk(const float* gsrc, unsigned lds_base,
                                                 int tid){
#pragma unroll
  for(int t = 0; t < 16; ++t){
    const unsigned off = (unsigned)(t*THREADS + tid) * 16u;   // 16 B per lane-op
    const unsigned dst = lds_base + off;
    asm volatile("global_load_async_to_lds_b128 %0, %1, %2"
                 :: "v"(dst), "v"(off), "s"(gsrc) : "memory");
  }
}

// ---- Tensor Data Mover staging (one instruction per 32 KB chunk) ---------
#if __has_builtin(__builtin_amdgcn_tensor_load_to_lds) && \
    __has_builtin(__builtin_amdgcn_s_wait_tensorcnt)
#define HAVE_TDM 1
typedef unsigned int v4u  __attribute__((ext_vector_type(4)));
typedef int          v8i_ __attribute__((ext_vector_type(8)));
typedef int          v4i_ __attribute__((ext_vector_type(4)));

__device__ __forceinline__ void tdm_load_32k(const float* gsrc, unsigned lds_base){
  const unsigned long long ga = (unsigned long long)(const void*)gsrc;
  // D# group 0: count=1 (valid), lds_addr, global_addr[56:0], type=2 ("image")
  v4u g0;
  g0[0] = 1u;
  g0[1] = lds_base;
  g0[2] = (unsigned)ga;
  g0[3] = ((unsigned)(ga >> 32) & 0x01FFFFFFu) | (2u << 30);
  // D# group 1: data_size=4B (code 2); 1-D tile of 8192 elements (= 32 KB)
  v8i_ g1;
  g1[0] = (int)(2u << 16);            // workgroup_mask=0, data_size=2 (4B)
  g1[1] = (int)(8192u << 16);         // tensor_dim0[15:0] in [31:16]
  g1[2] = (int)(1u << 16);            // tensor_dim0[31:16]=0, tensor_dim1=1
  g1[3] = (int)(8192u << 16);         // tensor_dim1 hi=0, tile_dim0=8192
  g1[4] = 1;                          // tile_dim1=1, tile_dim2=0
  g1[5] = 8192;                       // tensor_dim0_stride[31:0]
  g1[6] = 0;                          // stride hi / dim1_stride lo
  g1[7] = 0;
  v4i_ z4 = {0,0,0,0};
#if __clang_major__ >= 23
  v8i_ z8 = {0,0,0,0,0,0,0,0};
  __builtin_amdgcn_tensor_load_to_lds(g0, g1, z4, z4, z8, 0);
#else
  __builtin_amdgcn_tensor_load_to_lds(g0, g1, z4, z4, 0);
#endif
}
#endif

// raw fp32 (n, m, g) in LDS -> lwbf[slot*4+g][n][m] bf16 (packed pairs over m)
__device__ __forceinline__ void convert_w(const float* raw, unsigned short* lwbf,
                                          int slot, int tid){
  for(int u = tid; u < 16*64; u += THREADS){
    const int n = u >> 6, m2 = u & 63;
    const float4 a = *(const float4*)(raw + (n*CIN + 2*m2  )*4);
    const float4 b = *(const float4*)(raw + (n*CIN + 2*m2+1)*4);
    *(unsigned int*)(lwbf + ((slot*4+0)*16+n)*MPITCH + 2*m2) = pack_bf16x2(a.x, b.x);
    *(unsigned int*)(lwbf + ((slot*4+1)*16+n)*MPITCH + 2*m2) = pack_bf16x2(a.y, b.y);
    *(unsigned int*)(lwbf + ((slot*4+2)*16+n)*MPITCH + 2*m2) = pack_bf16x2(a.z, b.z);
    *(unsigned int*)(lwbf + ((slot*4+3)*16+n)*MPITCH + 2*m2) = pack_bf16x2(a.w, b.w);
  }
}

// stage NC (2 or 3) 16-channel chunks: async DMA pipelined against conversion
template<int NC>
__device__ __forceinline__ void stage_chunks(unsigned short* lwbf, float* raw0, float* raw1,
                                             const float* W, const int* nb, int tid){
  const unsigned r0 = lds_off(raw0), r1 = lds_off(raw1);
  async_copy_chunk(W + (size_t)nb[0]*512, r0, tid);
  async_copy_chunk(W + (size_t)nb[1]*512, r1, tid);
  WAIT_ASYNC(16);                 // chunk0 done (in-order retirement)
  __syncthreads();                // ...across all waves; also fences lwbf reuse
  convert_w(raw0, lwbf, 0, tid);
  WAIT_ASYNC(0);                  // chunk1 done
  __syncthreads();                // all waves done with raw0 + chunk1 visible
  if(NC > 2) async_copy_chunk(W + (size_t)nb[2]*512, r0, tid);
  convert_w(raw1, lwbf, 1, tid);
  if(NC > 2){
    WAIT_ASYNC(0);
    __syncthreads();
    convert_w(raw0, lwbf, 2, tid);
  }
  __syncthreads();                // fragments visible to all waves
}

#ifdef HAVE_TDM
// 3-chunk staging via the Tensor Data Mover; wave 0 drives the DMA engine.
__device__ __forceinline__ void stage_chunks_tdm3(unsigned short* lwbf, float* raw0,
                                                  float* raw1, const float* W,
                                                  const int* nb, int tid, int wv){
  const unsigned r0 = lds_off(raw0), r1 = lds_off(raw1);
  if(wv == 0){
    tdm_load_32k(W + (size_t)nb[0]*512, r0);
    tdm_load_32k(W + (size_t)nb[1]*512, r1);
    __builtin_amdgcn_s_wait_tensorcnt((short)1);   // chunk0 landed (in-order)
  }
  __syncthreads();
  convert_w(raw0, lwbf, 0, tid);
  if(wv == 0) __builtin_amdgcn_s_wait_tensorcnt((short)0);  // chunk1 landed
  __syncthreads();               // all waves done reading raw0 + chunk1 visible
  if(wv == 0) tdm_load_32k(W + (size_t)nb[2]*512, r0);
  convert_w(raw1, lwbf, 1, tid);
  if(wv == 0) __builtin_amdgcn_s_wait_tensorcnt((short)0);  // chunk2 landed
  __syncthreads();
  convert_w(raw0, lwbf, 2, tid);
  __syncthreads();               // fragments visible to all waves
}
#endif

// x (row, m, i) fp32 -> lx[i][row][m] bf16 (blade-major, packed pairs over m)
__device__ __forceinline__ void stage_x(unsigned short* lx, const float4* X4,
                                        int rowg0, int tid){
  for(int u = tid; u < BM*128; u += THREADS){
    const int rowl = u >> 7, rem = u & 127, m2 = rem >> 1, h = rem & 1;
    const float4 a = X4[((size_t)(rowg0+rowl)*CIN + 2*m2  )*2 + h];
    const float4 b = X4[((size_t)(rowg0+rowl)*CIN + 2*m2+1)*2 + h];
    const int i0 = 4*h;
    *(unsigned int*)(lx + ((i0+0)*BM + rowl)*MPITCH + 2*m2) = pack_bf16x2(a.x, b.x);
    *(unsigned int*)(lx + ((i0+1)*BM + rowl)*MPITCH + 2*m2) = pack_bf16x2(a.y, b.y);
    *(unsigned int*)(lx + ((i0+2)*BM + rowl)*MPITCH + 2*m2) = pack_bf16x2(a.z, b.z);
    *(unsigned int*)(lx + ((i0+3)*BM + rowl)*MPITCH + 2*m2) = pack_bf16x2(a.w, b.w);
  }
}

// ---- one 16x16 output tile x 8 blades: D = A(16x32) x B(32x16) over K=128 -
__device__ __forceinline__ void gemm_tensor(const Frag A[8][4], const unsigned short* lw,
                                            int slot, int ln, int half, float bias,
                                            v8f acc[8]){
#pragma unroll
  for(int i = 0; i < 8; ++i){
    const int g = bgrade(i);
    v8f c = {0.f,0.f,0.f,0.f,0.f,0.f,0.f,0.f};
#pragma unroll
    for(int kc = 0; kc < 4; ++kc){
      Frag B;
      const int base = ((slot*4+g)*16 + ln)*MPITCH + kc*32 + half*8;
      B.u[0] = *(const uint4*)(lw + base);
      B.u[1] = *(const uint4*)(lw + base + 16);
      c = __builtin_amdgcn_wmma_f32_16x16x32_bf16(false, A[i][kc].v, false, B.v,
                                                  (short)0, c, false, false);
    }
    acc[i] = c;
  }
#pragma unroll
  for(int r = 0; r < 8; ++r) acc[0][r] += bias;   // bias only on blade 0
}

// ---- epilogue pieces (per (lane,row) scalar octets) ----------------------
__device__ __forceinline__ void v_normalize(float V[8], const float sg[4]){
  const float vn0 = sqrtf(V[0]*V[0]);
  const float vn1 = sqrtf(V[1]*V[1] + V[2]*V[2] + V[3]*V[3]);
  const float vn2 = sqrtf(V[4]*V[4] + V[5]*V[5] + V[6]*V[6]);
  const float vn3 = sqrtf(V[7]*V[7]);
  const float d0 = 1.0f/(sg[0]*(vn0-1.0f)+1.0f + EPS_);
  const float d1 = 1.0f/(sg[1]*(vn1-1.0f)+1.0f + EPS_);
  const float d2 = 1.0f/(sg[2]*(vn2-1.0f)+1.0f + EPS_);
  const float d3 = 1.0f/(sg[3]*(vn3-1.0f)+1.0f + EPS_);
  V[0]*=d0; V[1]*=d1; V[2]*=d1; V[3]*=d1; V[4]*=d2; V[5]*=d2; V[6]*=d2; V[7]*=d3;
}

__device__ __forceinline__ void geo_product(const float K[8], const float V[8],
                                            const float gw[64], float kv[8]){
#pragma unroll
  for(int i = 0; i < 8; ++i){
#pragma unroll
    for(int k = 0; k < 8; ++k){
      const int   j = gpOut(i,k);                       // constant-folds
      const float s = gpSign(i,k);                      // constant-folds
      const float w = gw[bgrade(i)*16 + bgrade(j)*4 + bgrade(k)];
      kv[j] = fmaf(s * w * K[i], V[k], kv[j]);
    }
  }
}

// ---------------------------------------------------------------------------
__global__ void __launch_bounds__(THREADS, 1)
ce_block_kernel(const float* __restrict__ x,      const float* __restrict__ W_qkv,
                const float* __restrict__ b_qkv,  const float* __restrict__ silu_a,
                const float* __restrict__ silu_b, const float* __restrict__ gp_w,
                const float* __restrict__ norm_a, const float* __restrict__ ln_a,
                float* __restrict__ out){
  __shared__ __attribute__((aligned(16))) unsigned short smem[LX_HALFS];
  unsigned short* lx   = smem;                       // x tile (phase 0 only)
  unsigned short* lwbf = smem;                       // bf16 weight fragments
  float* raw0 = (float*)(smem + RAW0_H);             // 32 KB raw DMA buffers
  float* raw1 = (float*)(smem + RAW1_H);             // (alias the dead lx tail)

  const int tid  = threadIdx.x;
  const int wv   = tid >> 5;
  const int lane = tid & 31;
  const int ln   = lane & 15;
  const int half = lane >> 4;
  const int rowg0 = blockIdx.x * BM;

  // -- stage x, then pull all A fragments (8 blades x 4 K-chunks) to regs --
  stage_x(lx, (const float4*)x, rowg0, tid);
  __syncthreads();
  Frag af[8][4];
  {
    const int row = wv*16 + ln;
#pragma unroll
    for(int i = 0; i < 8; ++i)
#pragma unroll
      for(int kc = 0; kc < 4; ++kc){
        const int k0 = kc*32 + half*8;  // ISA 16-bit A 16x32 per-lane layout
        af[i][kc].u[0] = *(const uint4*)(lx + (i*BM + row)*MPITCH + k0);
        af[i][kc].u[1] = *(const uint4*)(lx + (i*BM + row)*MPITCH + k0 + 16);
      }
  }
  __syncthreads();   // lx dead; smem region becomes lwbf + raw buffers

  // -------- pass A: k,v -> geometric product -> per-row |kv| sums ---------
  float rs[8] = {0.f,0.f,0.f,0.f,0.f,0.f,0.f,0.f};
#pragma unroll 1
  for(int c = 0; c < 8; ++c){
    const int nbA[2] = { 128 + c*16, 256 + c*16 };          // k, v chunks
    stage_chunks<2>(lwbf, raw0, raw1, W_qkv, nbA, tid);

    const int nloc = c*16 + ln;
    v8f kacc[8], vacc[8];
    gemm_tensor(af, lwbf, 0, ln, half, b_qkv[128 + nloc], kacc);
    gemm_tensor(af, lwbf, 1, ln, half, b_qkv[256 + nloc], vacc);

    const float4 na = *(const float4*)(norm_a + nloc*4);
    const float sg[4] = { sigm(na.x), sigm(na.y), sigm(na.z), sigm(na.w) };
    float gw[64];
#pragma unroll
    for(int t = 0; t < 16; ++t)
      *(float4*)(gw + 4*t) = *(const float4*)(gp_w + (size_t)nloc*64 + 4*t);

#pragma unroll
    for(int r = 0; r < 8; ++r){
      float K[8], V[8];
#pragma unroll
      for(int i = 0; i < 8; ++i){ K[i] = kacc[i][r]; V[i] = vacc[i][r]; }
      v_normalize(V, sg);
      float kv[8] = {0.f,0.f,0.f,0.f,0.f,0.f,0.f,0.f};
      geo_product(K, V, gw, kv);
      float ss = 0.f;
#pragma unroll
      for(int j = 0; j < 8; ++j) ss = fmaf(kv[j], kv[j], ss);
      rs[r] += sqrtf(ss);
    }
  }

  // per-row reduction across the 16 lanes of each half (M = r + 8*half)
  float invn[8];
#pragma unroll
  for(int r = 0; r < 8; ++r){
    float s = rs[r];
    s += __shfl_xor(s, 1, 32);
    s += __shfl_xor(s, 2, 32);
    s += __shfl_xor(s, 4, 32);
    s += __shfl_xor(s, 8, 32);
    invn[r] = 1.0f / (s * (1.0f/COUT) + EPS_);
  }

  // -------- pass B: q,k,v -> gate, gp, scale, fused store -----------------
#pragma unroll 1
  for(int c = 0; c < 8; ++c){
    const int nbB[3] = { c*16, 128 + c*16, 256 + c*16 };    // q, k, v chunks
#ifdef HAVE_TDM
    stage_chunks_tdm3(lwbf, raw0, raw1, W_qkv, nbB, tid, wv);
#else
    stage_chunks<3>(lwbf, raw0, raw1, W_qkv, nbB, tid);
#endif

    const int nloc = c*16 + ln;
    v8f qacc[8], kacc[8], vacc[8];
    gemm_tensor(af, lwbf, 0, ln, half, b_qkv[      nloc], qacc);
    gemm_tensor(af, lwbf, 1, ln, half, b_qkv[128 + nloc], kacc);
    gemm_tensor(af, lwbf, 2, ln, half, b_qkv[256 + nloc], vacc);

    const float4 sa4 = *(const float4*)(silu_a + nloc*4);
    const float4 sb4 = *(const float4*)(silu_b + nloc*4);
    const float sa[4] = { sa4.x, sa4.y, sa4.z, sa4.w };
    const float sb[4] = { sb4.x, sb4.y, sb4.z, sb4.w };
    const float4 na = *(const float4*)(norm_a + nloc*4);
    const float sg[4] = { sigm(na.x), sigm(na.y), sigm(na.z), sigm(na.w) };
    const float lna = ln_a[nloc];
    float gw[64];
#pragma unroll
    for(int t = 0; t < 16; ++t)
      *(float4*)(gw + 4*t) = *(const float4*)(gp_w + (size_t)nloc*64 + 4*t);

#pragma unroll
    for(int r = 0; r < 8; ++r){
      float Q[8], K[8], V[8];
#pragma unroll
      for(int i = 0; i < 8; ++i){ Q[i] = qacc[i][r]; K[i] = kacc[i][r]; V[i] = vacc[i][r]; }
      // SiLU-style gate on q: qinv = [q0, |grade1|^2, |grade2|^2, |grade3|^2]
      const float qi[4] = { Q[0],
                            Q[1]*Q[1] + Q[2]*Q[2] + Q[3]*Q[3],
                            Q[4]*Q[4] + Q[5]*Q[5] + Q[6]*Q[6],
                            Q[7]*Q[7] };
#pragma unroll
      for(int i = 0; i < 8; ++i){
        const int g = bgrade(i);
        Q[i] *= sigm(fmaf(sa[g], qi[g], sb[g]));
      }
      v_normalize(V, sg);
      float kv[8] = {0.f,0.f,0.f,0.f,0.f,0.f,0.f,0.f};
      geo_product(K, V, gw, kv);

      const float kscale = lna * invn[r];
      float o[8];
#pragma unroll
      for(int j = 0; j < 8; ++j) o[j] = (Q[j] + kscale*kv[j]) * RSQRT2_;

      const size_t rowG = (size_t)rowg0 + wv*16 + half*8 + r;
      float4* dst = (float4*)(out + (rowG*COUT + nloc)*8);
      dst[0] = make_float4(o[0], o[1], o[2], o[3]);
      dst[1] = make_float4(o[4], o[5], o[6], o[7]);
    }
  }
}

// ---------------------------------------------------------------------------
extern "C" void kernel_launch(void* const* d_in, const int* in_sizes, int n_in,
                              void* d_out, int out_size, void* d_ws, size_t ws_size,
                              hipStream_t stream){
  (void)in_sizes; (void)n_in; (void)out_size; (void)d_ws; (void)ws_size;
  const float* x    = (const float*)d_in[0];
  const float* W    = (const float*)d_in[1];
  const float* bq   = (const float*)d_in[2];
  const float* sa   = (const float*)d_in[3];
  const float* sb   = (const float*)d_in[4];
  const float* gp   = (const float*)d_in[5];
  const float* na   = (const float*)d_in[6];
  const float* la   = (const float*)d_in[7];
  dim3 grid(B_TOTAL / BM);
  ce_block_kernel<<<grid, THREADS, 0, stream>>>(x, W, bq, sa, sb, gp, na, la,
                                                (float*)d_out);
}
